// AttentionBlock_15358803050771
// MI455X (gfx1250) — compile-verified, matrix operand order checked by reading, not run on hardware
//
#include <hip/hip_runtime.h>
#include <hip/hip_bf16.h>

typedef __bf16 bf16;
typedef __attribute__((ext_vector_type(16))) __bf16 v16bf;
typedef __attribute__((ext_vector_type(8)))  __bf16 v8bf;
typedef __attribute__((ext_vector_type(8)))  float  v8f;

#define N_B    4
#define N_H    8
#define N_T    2048
#define DMODEL 256

// ---------------------------------------------------------------------------
// Fragment loader for 16-bit WMMA operands (CDNA5 16x16x32 layout).
// Lane l (0-15):  row = l,    k = {0..7} and {16..23}
// Lane l (16-31): row = l-16, k = {8..15} and {24..31}
// ---------------------------------------------------------------------------
__device__ __forceinline__ v16bf load_frag_rm(const bf16* base, int ld) {
    int lane = threadIdx.x & 31;
    const bf16* p = base + (long)(lane & 15) * ld + ((lane >> 4) << 3);
    v8bf lo = *(const v8bf*)(p);
    v8bf hi = *(const v8bf*)(p + 16);
    return __builtin_shufflevector(lo, hi, 0,1,2,3,4,5,6,7,8,9,10,11,12,13,14,15);
}

__device__ __forceinline__ v8f wmma_bf16(v16bf a, v16bf b, v8f c) {
    return __builtin_amdgcn_wmma_f32_16x16x32_bf16(
        false, a, false, b, (short)0, c, false, false);
}

// ---------------------------------------------------------------------------
// CDNA5 async global->LDS copy (ASYNCcnt path), 16B per lane.
// ---------------------------------------------------------------------------
__device__ __forceinline__ unsigned lds_off(const void* p) {
    return (unsigned)(unsigned long long)p;
}
__device__ __forceinline__ void async_copy_b128(const void* gaddr, unsigned ldsaddr) {
    asm volatile("global_load_async_to_lds_b128 %0, %1, off"
                 :: "v"(ldsaddr), "v"((unsigned long long)gaddr)
                 : "memory");
}
__device__ __forceinline__ void wait_async_le8() {
    asm volatile("s_wait_asynccnt 0x8" ::: "memory");
}
__device__ __forceinline__ void wait_async0() {
    asm volatile("s_wait_asynccnt 0x0" ::: "memory");
}

// ---------------------------------------------------------------------------
// Converters (f32 -> bf16); Q transposed so GEMM B-operands are K-contiguous.
// ---------------------------------------------------------------------------
__global__ void k_cvt(const float* __restrict__ in, bf16* __restrict__ out, int n) {
    int i = blockIdx.x * blockDim.x + threadIdx.x;
    if (i < n) out[i] = (bf16)in[i];
}

__global__ void k_cvt_qt(const float* __restrict__ Q, bf16* __restrict__ Qt) {
    int idx = blockIdx.x * blockDim.x + threadIdx.x;     // H*256*256
    int j = idx & 255, i = (idx >> 8) & 255, h = idx >> 16;
    Qt[(h << 16) + (j << 8) + i] = (bf16)Q[idx];
}

// ---------------------------------------------------------------------------
// bf16 GEMM:  C(MxN, rm) = A(MxK, rm) * Bt(NxK, rm)^T, f32 accum. K == 256.
// Software-pipelined: WMMAs for the current half-k-step run against
// fragments loaded one stage earlier; next-stage loads are in flight.
// ---------------------------------------------------------------------------
__global__ __launch_bounds__(256) __attribute__((amdgpu_waves_per_eu(1)))
void k_gemm_nt(const bf16* __restrict__ A, const bf16* __restrict__ Bt,
               bf16* __restrict__ C, int ldc, int nBlk,
               long aOffB, long aOffH, long bOffB, long bOffH,
               long cOffB, long cOffH)
{
    const int K = DMODEL, KC = K >> 5;                // 8 k-steps of 32
    int y = blockIdx.y;
    int b = y >> 3, h = y & 7;
    const bf16* Ap = A  + b * aOffB + h * aOffH;
    const bf16* Bp = Bt + b * bOffB + h * bOffH;
    bf16*       Cp = C  + b * cOffB + h * cOffH;

    int mb = blockIdx.x / nBlk, nb = blockIdx.x % nBlk;
    int wid = threadIdx.x >> 5, lane = threadIdx.x & 31;
    int row0 = mb * 128 + wid * 16;
    int col0 = nb * 256;

    const bf16* arow = Ap + (long)row0 * K;
    #define BPTR(nt, kc) (Bp + (long)(col0 + (nt) * 16) * K + (kc) * 32)

    v8f acc[16];
    v8f z = {};
    for (int i = 0; i < 16; ++i) acc[i] = z;

    v16bf a_cur = load_frag_rm(arow, K);
    v16bf bs_cur[8];
    #pragma unroll
    for (int j = 0; j < 8; ++j) bs_cur[j] = load_frag_rm(BPTR(j, 0), K);

    for (int kc = 0; kc < KC; ++kc) {
        // stage 1: load half1 of this k-step, compute half0
        v16bf bs_next[8];
        #pragma unroll
        for (int j = 0; j < 8; ++j) bs_next[j] = load_frag_rm(BPTR(8 + j, kc), K);
        #pragma unroll
        for (int j = 0; j < 8; ++j) acc[j] = wmma_bf16(a_cur, bs_cur[j], acc[j]);

        // stage 2: load half0 + A of next k-step, compute half1
        v16bf a_next = a_cur;
        v16bf bs2[8];
        #pragma unroll
        for (int j = 0; j < 8; ++j) bs2[j] = bs_cur[j];
        if (kc + 1 < KC) {
            a_next = load_frag_rm(arow + (kc + 1) * 32, K);
            #pragma unroll
            for (int j = 0; j < 8; ++j) bs2[j] = load_frag_rm(BPTR(j, kc + 1), K);
        }
        #pragma unroll
        for (int j = 0; j < 8; ++j) acc[8 + j] = wmma_bf16(a_cur, bs_next[j], acc[8 + j]);

        a_cur = a_next;
        #pragma unroll
        for (int j = 0; j < 8; ++j) bs_cur[j] = bs2[j];
    }
    #undef BPTR

    int n = lane & 15, mo = (lane >> 4) << 3;
    for (int nt = 0; nt < 16; ++nt)
        for (int r = 0; r < 8; ++r)
            Cp[(long)(row0 + mo + r) * ldc + col0 + nt * 16 + n] = (bf16)acc[nt][r];
}

// ---------------------------------------------------------------------------
// Fused kernel: out[b,t,:] += sum_h tril(P R^T)[t,u] * V[u,:]
// grid (16,8,4), 8 waves. P A-fragments in registers; R/V chunks double-
// buffered in LDS via async copies so the DMA for chunk ub+1 overlaps the
// WMMAs of chunk ub.
// ---------------------------------------------------------------------------
__global__ __launch_bounds__(256) __attribute__((amdgpu_waves_per_eu(1)))
void k_attn(const bf16* __restrict__ P, const bf16* __restrict__ R,
            const bf16* __restrict__ Vt, float* __restrict__ out)
{
    int tb = blockIdx.x, h = blockIdx.y, b = blockIdx.z;
    const bf16* Pb = P  + (long)(b * N_H + h) * N_T * DMODEL;
    const bf16* Rb = R  + (long)b * N_T * DMODEL;
    const bf16* Vb = Vt + (long)(b * N_H + h) * DMODEL * N_T;
    float*    outb = out + (long)b * N_T * DMODEL;

    int tid  = threadIdx.x;
    int wid  = tid >> 5, lane = tid & 31;
    int t0   = tb * 128 + wid * 16;

    __shared__ bf16 sR[2][32 * DMODEL];   // R chunk double buffer  (2 x 16 KB)
    __shared__ bf16 sV[2][DMODEL * 32];   // Vt chunk double buffer (2 x 16 KB)
    __shared__ bf16 sS[8][16 * 32];       // per-wave S scratch          (8 KB)
    bf16* sw = &sS[wid][0];

    // invariant A-fragments: this wave's 16 P-rows, full K=256
    v16bf a1[8];
    #pragma unroll
    for (int kc = 0; kc < 8; ++kc)
        a1[kc] = load_frag_rm(Pb + (long)t0 * DMODEL + kc * 32, DMODEL);

    v8f acc[16];
    v8f z = {};
    for (int i = 0; i < 16; ++i) acc[i] = z;

    // 8 async b128 per thread per chunk: 4 for R (contiguous 16 KB), 4 for V
    auto stage = [&](int ub, int buf) {
        const bf16* rsrc = Rb + (long)(ub << 5) * DMODEL;
        #pragma unroll
        for (int i = 0; i < 4; ++i) {
            int e = tid * 8 + i * 2048;
            async_copy_b128(rsrc + e, lds_off(&sR[buf][e]));
        }
        const bf16* vsrc = Vb + (long)tid * N_T + (ub << 5);
        #pragma unroll
        for (int i = 0; i < 4; ++i)
            async_copy_b128(vsrc + i * 8, lds_off(&sV[buf][tid * 32 + i * 8]));
    };

    int nUB = ((tb * 128 + 127) >> 5) + 1;            // uniform per block
    stage(0, 0);
    for (int ub = 0; ub < nUB; ++ub) {
        int cur = ub & 1;
        bool active = ((ub << 5) <= t0 + 15);
        int u0 = ub << 5;

        if (ub + 1 < nUB) {                            // prefetch next chunk
            stage(ub + 1, cur ^ 1);
            wait_async_le8();                          // in-order: chunk ub done
        } else {
            wait_async0();
        }
        __syncthreads();                               // staged data visible

        if (active) {
            // S(16x32) = P_tile @ R_chunk^T  (A regs, B from LDS)
            const bf16* sRc = &sR[cur][0];
            v8f s0 = z, s1 = z;
            #pragma unroll
            for (int kc = 0; kc < 8; ++kc) {
                v16bf b0 = load_frag_rm(sRc + kc * 32, DMODEL);
                v16bf b1 = load_frag_rm(sRc + 16 * DMODEL + kc * 32, DMODEL);
                s0 = wmma_bf16(a1[kc], b0, s0);
                s1 = wmma_bf16(a1[kc], b1, s1);
            }
            // causal mask on diagonal-straddling chunks
            if (u0 + 31 > t0) {
                int mrow = t0 + ((lane >> 4) << 3);
                int nlo  = u0 + (lane & 15);
                #pragma unroll
                for (int r = 0; r < 8; ++r) {
                    int t = mrow + r;
                    if (nlo > t)      s0[r] = 0.0f;
                    if (nlo + 16 > t) s1[r] = 0.0f;
                }
            }
            // C-layout -> A-layout via wave-private LDS (DS ops are in-order
            // within a wave, so no workgroup barrier is needed here)
            int n = lane & 15, mo = (lane >> 4) << 3;
            #pragma unroll
            for (int r = 0; r < 8; ++r) {
                sw[(mo + r) * 32 + n]      = (bf16)s0[r];
                sw[(mo + r) * 32 + n + 16] = (bf16)s1[r];
            }
            v16bf a2 = load_frag_rm(sw, 32);

            // O += S @ V chunk, ds fragments preloaded in groups of 8
            const bf16* sVc = &sV[cur][0];
            #pragma unroll
            for (int half = 0; half < 2; ++half) {
                v16bf bs[8];
                #pragma unroll
                for (int j = 0; j < 8; ++j)
                    bs[j] = load_frag_rm(sVc + (half * 8 + j) * 16 * 32, 32);
                #pragma unroll
                for (int j = 0; j < 8; ++j)
                    acc[half * 8 + j] = wmma_bf16(a2, bs[j], acc[half * 8 + j]);
            }
        }
        __syncthreads();                               // chunk fully consumed
    }

    // head reduction into f32 output (native f32 atomics)
    int n = lane & 15, mo = (lane >> 4) << 3;
    for (int nt = 0; nt < 16; ++nt)
        for (int r = 0; r < 8; ++r)
            unsafeAtomicAdd(&outb[(long)(t0 + mo + r) * DMODEL + nt * 16 + n],
                            acc[nt][r]);
}

// ---------------------------------------------------------------------------
extern "C" void kernel_launch(void* const* d_in, const int* in_sizes, int n_in,
                              void* d_out, int out_size, void* d_ws, size_t ws_size,
                              hipStream_t stream) {
    const float* r_prime = (const float*)d_in[0];   // [1,4,2048,256]
    const float* E       = (const float*)d_in[1];   // [1,8,256,256]
    const float* Q       = (const float*)d_in[2];   // [1,8,256,256]
    float* out = (float*)d_out;                     // [1,4,2048,256]

    char* ws = (char*)d_ws;
    size_t off = 0;
    bf16* Rbf = (bf16*)(ws + off); off += (size_t)N_B * N_T * DMODEL * 2;          // 4 MB
    bf16* Qt  = (bf16*)(ws + off); off += (size_t)N_H * DMODEL * DMODEL * 2;       // 1 MB
    bf16* Ebf = (bf16*)(ws + off); off += (size_t)N_H * DMODEL * DMODEL * 2;       // 1 MB
    bf16* Pbf = (bf16*)(ws + off); off += (size_t)N_B * N_H * N_T * DMODEL * 2;    // 32 MB
    bf16* Vtb = (bf16*)(ws + off); off += (size_t)N_B * N_H * DMODEL * N_T * 2;    // 32 MB

    {
        int n = N_B * N_T * DMODEL;
        k_cvt<<<(n + 255) / 256, 256, 0, stream>>>(r_prime, Rbf, n);
    }
    {
        int n = N_H * DMODEL * DMODEL;
        k_cvt_qt<<<(n + 255) / 256, 256, 0, stream>>>(Q, Qt);
        k_cvt<<<(n + 255) / 256, 256, 0, stream>>>(E, Ebf, n);
    }

    // P[b,h] = R[b] @ Q[h]   (M=2048, N=256, K=256), Bt = Qt
    k_gemm_nt<<<dim3((N_T / 128) * 1, N_B * N_H), 256, 0, stream>>>(
        Rbf, Qt, Pbf, DMODEL, 1,
        (long)N_T * DMODEL, 0,
        0, (long)DMODEL * DMODEL,
        (long)N_H * N_T * DMODEL, (long)N_T * DMODEL);

    // Vt[b,h] = E[h] @ R[b]^T (M=256, N=2048, K=256), Bt = Rbf
    k_gemm_nt<<<dim3((DMODEL / 128) * (N_T / 256), N_B * N_H), 256, 0, stream>>>(
        Ebf, Rbf, Vtb, N_T, N_T / 256,
        0, (long)DMODEL * DMODEL,
        (long)N_T * DMODEL, 0,
        (long)N_H * DMODEL * N_T, (long)DMODEL * N_T);

    hipMemsetAsync(d_out, 0, (size_t)out_size * sizeof(float), stream);
    k_attn<<<dim3(N_T / 128, N_H, N_B), 256, 0, stream>>>(Pbf, Rbf, Vtb, out);
}
